// Conv2d_22694607192245
// MI455X (gfx1250) — compile-verified, matrix operand order checked by reading, not run on hardware
//
#include <hip/hip_runtime.h>

// ---- problem constants (from reference) ----
#define BATCH 32
#define C_IN  64
#define O_CH  128
#define HH    56
#define WW    56
#define HWSZ  3136              // 56*56
#define KTOT  576               // C_IN * 9 ; reordered as k = tap*64 + c
#define BN    128               // N tile per block (flattened over batch)
#define BK    32                // K step (one tap, half the channels)
#define NSTEPS 18               // KTOT / BK  (9 taps x 2 channel halves)
#define NTOT  (BATCH * HWSZ)    // 100352
#define NBLK  (NTOT / BN)       // 784

// LDS row strides in halves
#define LDA 584                 // 576 + 8 pad : 1168B row, 16B aligned, conflict-free b128
#define LDB 40                  // 32 + 8 pad  : 80B row

typedef __attribute__((ext_vector_type(16))) __bf16        v16bf;
typedef __attribute__((ext_vector_type(8)))  float         v8f;
typedef __attribute__((ext_vector_type(4)))  unsigned int  v4u;

union FragCast { v4u u[2]; v16bf v; };   // trivial -> valid union

// hardware bf16 convert (RNE) + pack
__device__ __forceinline__ unsigned int pack_bf16(float a, float b) {
  __bf16 x = (__bf16)a, y = (__bf16)b;
  unsigned int ux = (unsigned int)__builtin_bit_cast(unsigned short, x);
  unsigned int uy = (unsigned int)__builtin_bit_cast(unsigned short, y);
  return ux | (uy << 16);
}

// Stage one K-step of the im2col B tile: [BN x BK] bf16, stored [n][k].
// K-step ks = (tap, chalf): tap is wave-uniform -> SALU math only.
// Branchless padding: clamp coords (address always valid), unconditional
// strided loads, then v_cndmask the value to zero.
__device__ __forceinline__ void stage_B(int ks, unsigned short* __restrict__ Bt,
                                        const float* __restrict__ xb,
                                        int i, int j, int t) {
  const int n   = t & 127;
  const int kq  = (t >> 7) & 1;          // which 16-channel half this thread loads
  const int tap = ks >> 1;               // uniform -> SALU
  const int ch0 = (ks & 1) * 32 + kq * 16;
  const int kh  = tap / 3;               // uniform small div -> SALU
  const int kw  = tap - kh * 3;
  const int ih  = i + kh - 1;
  const int iw  = j + kw - 1;
  const bool valid = ((unsigned)ih < (unsigned)HH) && ((unsigned)iw < (unsigned)WW);
  const int ihc = ih < 0 ? 0 : (ih > HH - 1 ? HH - 1 : ih);   // v_med3-able clamp
  const int iwc = iw < 0 ? 0 : (iw > WW - 1 ? WW - 1 : iw);
  const float* p = xb + ch0 * HWSZ + ihc * WW + iwc;           // always in-bounds

  float v[16];
#pragma unroll
  for (int cc = 0; cc < 16; ++cc)        // unconditional coalesced loads
    v[cc] = p[cc * HWSZ];
#pragma unroll
  for (int cc = 0; cc < 16; ++cc)        // select -> v_cndmask, no branches
    v[cc] = valid ? v[cc] : 0.0f;

  v4u s0, s1;
  s0.x = pack_bf16(v[0],  v[1]);  s0.y = pack_bf16(v[2],  v[3]);
  s0.z = pack_bf16(v[4],  v[5]);  s0.w = pack_bf16(v[6],  v[7]);
  s1.x = pack_bf16(v[8],  v[9]);  s1.y = pack_bf16(v[10], v[11]);
  s1.z = pack_bf16(v[12], v[13]); s1.w = pack_bf16(v[14], v[15]);
  v4u* d = (v4u*)(Bt + n * LDB + kq * 16);   // bytes: n*80 + kq*32, 16B aligned
  d[0] = s0;
  d[1] = s1;
}

__global__ void __launch_bounds__(256)
conv3x3_implicit_gemm_bf16_wmma(const float* __restrict__ x,
                                const float* __restrict__ flt,
                                float* __restrict__ out) {
  // CDNA5: 320KB LDS per WGP -> keep the ENTIRE bf16 filter matrix resident.
  __shared__ __align__(16) unsigned short lA[O_CH * LDA];     // ~146 KB
  __shared__ __align__(16) unsigned short lB[2][BN * LDB];    //  20 KB

  const int tile = blockIdx.x;          // 0..783
  const int t    = threadIdx.x;
  const int lane = t & 31;
  const int wave = t >> 5;              // 0..7
  const int mq   = wave & 3;            // M group: 32 rows each
  const int nq   = wave >> 2;           // N group: 64 cols each
  const int hsel = (lane >> 4) & 1;
  const int l15  = lane & 15;

  // per-thread im2col coordinates for B staging (computed once)
  const int n_st = t & 127;
  const int ng   = tile * BN + n_st;
  const int b_st = ng / HWSZ;
  const int hw   = ng - b_st * HWSZ;
  const int i_st = hw / WW;
  const int j_st = hw - i_st * WW;
  const float* xb = x + (size_t)b_st * (C_IN * HWSZ);

  // ---- one-time: all filters -> LDS bf16, reordered k = tap*64 + c ----
  {
    const int o    = t >> 1;
    const int half = t & 1;                       // k range half*288 .. +287
    const float* fo = flt + o * KTOT;
    unsigned short* ao = lA + o * LDA + half * 288;
#pragma unroll 4
    for (int chunk = 0; chunk < 36; ++chunk) {
      unsigned int u[4];
#pragma unroll
      for (int q = 0; q < 4; ++q) {
        const int k0  = half * 288 + chunk * 8 + q * 2;  // even -> c even, same tap
        const int tap = k0 >> 6;
        const int c0  = k0 & 63;
        u[q] = pack_bf16(fo[c0 * 9 + tap], fo[(c0 + 1) * 9 + tap]);
      }
      v4u s; s.x = u[0]; s.y = u[1]; s.z = u[2]; s.w = u[3];
      *(v4u*)(ao + chunk * 8) = s;
    }
  }

  stage_B(0, &lB[0][0], xb, i_st, j_st, t);

  v8f acc[2][4] = {};                   // 2 M-frags x 4 N-frags of 16x16 f32

#pragma unroll 2
  for (int ks = 0; ks < NSTEPS; ++ks) {
    __syncthreads();                    // A-init + B[cur] visible; prev compute done
    const int cur = ks & 1;
    if (ks + 1 < NSTEPS)
      stage_B(ks + 1, &lB[cur ^ 1][0], xb, i_st, j_st, t);

    // ---- preload ALL fragments, then issue 8 WMMAs back-to-back ----
    FragCast fa[2], fb[4];
#pragma unroll
    for (int mf = 0; mf < 2; ++mf) {
      const unsigned short* pa =
          &lA[(mq * 32 + mf * 16 + l15) * LDA + ks * 32 + hsel * 8];
      fa[mf].u[0] = *(const v4u*)(pa);
      fa[mf].u[1] = *(const v4u*)(pa + 16);
    }
#pragma unroll
    for (int ns = 0; ns < 4; ++ns) {
      const unsigned short* pb =
          &lB[cur][(nq * 64 + ns * 16 + l15) * LDB + hsel * 8];
      fb[ns].u[0] = *(const v4u*)(pb);
      fb[ns].u[1] = *(const v4u*)(pb + 16);
    }
#pragma unroll
    for (int ns = 0; ns < 4; ++ns) {
      acc[0][ns] = __builtin_amdgcn_wmma_f32_16x16x32_bf16(
          false, fa[0].v, false, fb[ns].v, (short)0, acc[0][ns], false, false);
      acc[1][ns] = __builtin_amdgcn_wmma_f32_16x16x32_bf16(
          false, fa[1].v, false, fb[ns].v, (short)0, acc[1][ns], false, false);
    }
  }

  // ---- epilogue: 16-wide N groups never straddle a batch (3136 = 16*196) ----
#pragma unroll
  for (int ns = 0; ns < 4; ++ns) {
    const int n   = tile * BN + nq * 64 + ns * 16 + l15;
    const int b   = n / HWSZ;
    const int hwn = n - b * HWSZ;
    float* ob = out + (size_t)b * (O_CH * HWSZ) + hwn;
#pragma unroll
    for (int mf = 0; mf < 2; ++mf) {
      const int obase = mq * 32 + mf * 16 + hsel * 8;
#pragma unroll
      for (int r = 0; r < 8; ++r)
        ob[(obase + r) * HWSZ] = acc[mf][ns][r];
    }
  }
}

extern "C" void kernel_launch(void* const* d_in, const int* in_sizes, int n_in,
                              void* d_out, int out_size, void* d_ws, size_t ws_size,
                              hipStream_t stream) {
  (void)in_sizes; (void)n_in; (void)out_size; (void)d_ws; (void)ws_size;
  const float* x   = (const float*)d_in[0];   // [32,64,56,56]
  const float* flt = (const float*)d_in[1];   // [128,64,3,3]
  float* out = (float*)d_out;                 // [32,128,56,56]
  conv3x3_implicit_gemm_bf16_wmma<<<dim3(NBLK), dim3(256), 0, stream>>>(x, flt, out);
}
